// CoCColumnParallelLinearImpl_44856638439890
// MI455X (gfx1250) — matrix operand order repaired and checked by reading, compiled, not run on hardware
//
#include <hip/hip_runtime.h>

typedef __attribute__((ext_vector_type(16))) __bf16 v16bf;
typedef __attribute__((ext_vector_type(8)))  __bf16 v8bf;
typedef __attribute__((ext_vector_type(4)))  __bf16 v4bf;
typedef __attribute__((ext_vector_type(8)))  float  v8f;

namespace {
constexpr int Mdim = 8192;   // S*B
constexpr int Kdim = 4096;   // H
constexpr int Ndim = 16384;  // F
constexpr int TM = 128, TN = 128, TK = 32;
constexpr int NK = Kdim / TK;
constexpr int LDSS = 40;     // halves per LDS row: 32 data + 8 pad (16B-aligned, conflict-free)

__device__ __forceinline__ v8f wmma_bf16(v16bf a, v16bf b, v8f c) {
  return __builtin_amdgcn_wmma_f32_16x16x32_bf16(false, a, false, b, (short)0, c,
                                                 false, false);
}

// Two 16-byte LDS loads concatenated into one 16-element bf16 fragment.
__device__ __forceinline__ v16bf ldfrag(const __bf16* p, int delta_halves) {
  v8bf lo = *(const v8bf*)p;
  v8bf hi = *(const v8bf*)(p + delta_halves);
  return __builtin_shufflevector(lo, hi, 0, 1, 2, 3, 4, 5, 6, 7,
                                 8, 9, 10, 11, 12, 13, 14, 15);
}

// fp32 -> bf16 hi + bf16 lo residual (x ~= hi + lo, ~16 mantissa bits kept)
__device__ __forceinline__ void split4(float4 x, v4bf& h, v4bf& l) {
  float xs[4] = {x.x, x.y, x.z, x.w};
#pragma unroll
  for (int i = 0; i < 4; ++i) {
    __bf16 hb = (__bf16)xs[i];
    h[i] = hb;
    l[i] = (__bf16)(xs[i] - (float)hb);
  }
}
}  // namespace

__global__ void __launch_bounds__(256)
coc_column_parallel_linear_bf16x3(const float* __restrict__ A,
                                  const float* __restrict__ W,
                                  const float* __restrict__ bias,
                                  float* __restrict__ out) {
  // [stage][plane: A_hi, A_lo, B_hi, B_lo][128 rows * 40 halves]  = 80 KB
  __shared__ __align__(16) __bf16 lds[2][4][TM * LDSS];

  const int tid   = threadIdx.x;
  const int lane  = tid & 31;
  const int wave  = tid >> 5;
  const int waveM = wave & 1;   // 0..1 -> 64-row slab
  const int waveN = wave >> 1;  // 0..3 -> 32-col slab
  const int l16   = lane & 15;
  const int khalf = lane >> 4;  // 0 or 1: which K half-group this lane holds

  const int n0 = blockIdx.x * TN;
  const int m0 = blockIdx.y * TM;

  // Cooperative fill coords: 4 passes, 32 rows/pass, 8 threads/row (float4 each)
  const int frow = tid >> 3;       // 0..31
  const int fcol = (tid & 7) * 4;  // 0,4,...,28

  v8f acc[4][2];
  const v8f vzero = {0.f, 0.f, 0.f, 0.f, 0.f, 0.f, 0.f, 0.f};
#pragma unroll
  for (int t = 0; t < 4; ++t)
#pragma unroll
    for (int u = 0; u < 2; ++u) acc[t][u] = vzero;

  auto gload = [&](float4* pa, float4* pb, int kt) {
    const float* abase = A + (size_t)m0 * Kdim + (size_t)kt * TK + fcol;
    const float* wbase = W + (size_t)n0 * Kdim + (size_t)kt * TK + fcol;
#pragma unroll
    for (int q = 0; q < 4; ++q) {
      const int r = q * 32 + frow;
      pa[q] = *(const float4*)(abase + (size_t)r * Kdim);
      pb[q] = *(const float4*)(wbase + (size_t)r * Kdim);
    }
  };

  auto lstore = [&](int st, const float4* pa, const float4* pb) {
#pragma unroll
    for (int q = 0; q < 4; ++q) {
      const int off = (q * 32 + frow) * LDSS + fcol;
      v4bf h, l;
      split4(pa[q], h, l);
      *(v4bf*)&lds[st][0][off] = h;
      *(v4bf*)&lds[st][1][off] = l;
      split4(pb[q], h, l);
      *(v4bf*)&lds[st][2][off] = h;
      *(v4bf*)&lds[st][3][off] = l;
    }
  };

  auto compute = [&](int st) {
    const __bf16* Ah = lds[st][0];
    const __bf16* Al = lds[st][1];
    const __bf16* Bh = lds[st][2];
    const __bf16* Bl = lds[st][3];
    // ISA 7.12.2: 16-bit A 16x32 -> lane<16 holds K {0..7,16..23}, lane>=16 K {8..15,24..31}
    const int akb = khalf * 8;
    // 16-bit B 32x16 -> lane holds col lane%16, same K set as A: {0..7,16..23} / {8..15,24..31}
    const int bkb = khalf * 8;

    v16bf bh[2], bl[2];
#pragma unroll
    for (int u = 0; u < 2; ++u) {
      const int col = waveN * 32 + u * 16 + l16;
      bh[u] = ldfrag(Bh + col * LDSS + bkb, 16);
      bl[u] = ldfrag(Bl + col * LDSS + bkb, 16);
    }
#pragma unroll
    for (int t = 0; t < 4; ++t) {
      const int row = waveM * 64 + t * 16 + l16;
      v16bf ah = ldfrag(Ah + row * LDSS + akb, 16);
      v16bf al = ldfrag(Al + row * LDSS + akb, 16);
#pragma unroll
      for (int u = 0; u < 2; ++u) {
        // bf16x3 split product: hi*lo + lo*hi + hi*hi (lo*lo dropped, ~2^-17 rel err)
        acc[t][u] = wmma_bf16(ah, bl[u], acc[t][u]);
        acc[t][u] = wmma_bf16(al, bh[u], acc[t][u]);
        acc[t][u] = wmma_bf16(ah, bh[u], acc[t][u]);
      }
    }
  };

  float4 pa[4], pb[4];
  gload(pa, pb, 0);
  lstore(0, pa, pb);
  __syncthreads();

  for (int kt = 0; kt < NK; ++kt) {
    const int st = kt & 1;
    if (kt + 1 < NK) gload(pa, pb, kt + 1);  // prefetch next K-slab during WMMA
    compute(st);
    if (kt + 1 < NK) lstore(st ^ 1, pa, pb);
    __syncthreads();
  }

  // Epilogue: C/D layout — VGPR j: lanes 0-15 -> M=j, lanes 16-31 -> M=j+8; N=lane%16
#pragma unroll
  for (int u = 0; u < 2; ++u) {
    const int n = n0 + waveN * 32 + u * 16 + l16;
    const float bv = bias[n];
#pragma unroll
    for (int t = 0; t < 4; ++t) {
      const int mb = m0 + waveM * 64 + t * 16 + khalf * 8;
      float* o = out + (size_t)mb * Ndim + n;
#pragma unroll
      for (int j = 0; j < 8; ++j) o[(size_t)j * Ndim] = acc[t][u][j] + bv;
    }
  }
}

extern "C" void kernel_launch(void* const* d_in, const int* in_sizes, int n_in,
                              void* d_out, int out_size, void* d_ws, size_t ws_size,
                              hipStream_t stream) {
  const float* input  = (const float*)d_in[0];  // [4096, 2, 4096] fp32
  const float* weight = (const float*)d_in[1];  // [16384, 4096] fp32
  const float* bias   = (const float*)d_in[2];  // [16384] fp32
  float* out = (float*)d_out;                   // [4096, 2, 16384] fp32

  dim3 grid(Ndim / TN, Mdim / TM);  // (128, 64)
  dim3 block(256);                  // 8 wave32s
  coc_column_parallel_linear_bf16x3<<<grid, block, 0, stream>>>(input, weight, bias, out);
}
